// TimeBlock_42683384987781
// MI455X (gfx1250) — matrix operand order repaired and checked by reading, not compiled, not run on hardware
//
#include <hip/hip_runtime.h>
#include <math.h>

#define NN 4096
#define DD 256

typedef __bf16 bf16_t;
typedef bf16_t v8bf  __attribute__((ext_vector_type(8)));
typedef bf16_t v16bf __attribute__((ext_vector_type(16)));
typedef float  v8f   __attribute__((ext_vector_type(8)));

__device__ __forceinline__ bf16_t f2bf(float x) {
  union { float f; unsigned u; } c; c.f = x;
  unsigned r = c.u + 0x7FFFu + ((c.u >> 16) & 1u);   // round-to-nearest-even
  unsigned short h = (unsigned short)(r >> 16);
  return __builtin_bit_cast(bf16_t, h);
}

__device__ __forceinline__ v16bf cat16(v8bf lo, v8bf hi) {
  return __builtin_shufflevector(lo, hi, 0,1,2,3,4,5,6,7,8,9,10,11,12,13,14,15);
}

__device__ __forceinline__ v8f wmma_bf16(v16bf a, v16bf b, v8f c) {
  return __builtin_amdgcn_wmma_f32_16x16x32_bf16(false, a, false, b, (short)0, c,
                                                 false, false);
}

// ---------------------------------------------------------------------------
// K1: f32 -> bf16 conversions (row-major cur, transposed cur/pre), zero scalars
// ---------------------------------------------------------------------------
__global__ void prep_kernel(const float* __restrict__ cur, const float* __restrict__ pre,
                            bf16_t* __restrict__ curBf, bf16_t* __restrict__ curT,
                            bf16_t* __restrict__ preT, float* __restrict__ scal) {
  int idx = blockIdx.x * blockDim.x + threadIdx.x;
  if (idx < 8) scal[idx] = 0.0f;
  if (idx >= NN * DD) return;
  int i = idx >> 8, d = idx & 255;
  float c = cur[idx], p = pre[idx];
  curBf[idx] = f2bf(c);
  curT[(size_t)d * NN + i] = f2bf(c);
  preT[(size_t)d * NN + i] = f2bf(p);
}

// ---------------------------------------------------------------------------
// Fused big GEMM over an incidence matrix (templated on SPATIAL scatter):
//   diff2  += | A_raw @ B - master |^2   (fused Frobenius diff, no recon buffer)
//   fea     = relu(A) @ B + cur
//   deg     = rowsum(relu(A)) + 1
//   scal[abs] += sum|A|, scal[sq] += sum A^2
// BM=32 rows, full 256 cols, BK=64, 256 threads (8 waves).
// ---------------------------------------------------------------------------
template <int SPATIAL>
__global__ __launch_bounds__(256) void big_fused(
    const float* __restrict__ inc,
    const bf16_t* __restrict__ Bt,           // [256][4096] bf16 (col-major B)
    const float* __restrict__ curF,
    const float* __restrict__ master,
    float* __restrict__ fea,
    float* __restrict__ deg, float* __restrict__ scal) {
  __shared__ __align__(16) bf16_t lAr[32][72];
  __shared__ __align__(16) bf16_t lAl[32][72];
  __shared__ __align__(16) bf16_t lBt[256][72];
  __shared__ float degp[32][8];
  __shared__ float red1[256];
  __shared__ float red2[256];
  __shared__ float red3[256];

  const int tid = threadIdx.x;
  const int lane = tid & 31, wave = tid >> 5;
  const int rowBase = blockIdx.x * 32;
  const int lrow = tid >> 3;                // loader row 0..31
  const int kseg = (tid & 7) * 8;           // loader k start within BK
  const int gi = rowBase + lrow;

  const int mbase = (wave & 1) * 16;        // wave's 16-row tile
  const int nbase = (wave >> 1) * 64;       // wave's 64-col strip
  const int mr = lane & 15;
  const int half = lane >> 4;

  v8f zero = {0.f,0.f,0.f,0.f,0.f,0.f,0.f,0.f};
  v8f accR[4], accL[4];
#pragma unroll
  for (int t = 0; t < 4; ++t) { accR[t] = zero; accL[t] = zero; }

  float absP = 0.f, sqP = 0.f, relP = 0.f;

  for (int k0 = 0; k0 < NN; k0 += 64) {
    // ---- A tiles: raw + relu, branchless diagonal handling ----
    float vals[8];
    if (SPATIAL) {
      const float* rowp = inc + (size_t)gi * (NN - 1);
#pragma unroll
      for (int e = 0; e < 8; ++e) {
        int j = k0 + kseg + e;
        int idx = j - (j > gi ? 1 : 0);     // valid address even when j==gi
        float v = rowp[idx];
        vals[e] = (j == gi) ? 0.f : v;      // v_cndmask, no branch
      }
      if (k0 + 64 < NN)
        __builtin_prefetch(rowp + k0 + kseg + 64, 0, 1);
    } else {
      const float4* p = reinterpret_cast<const float4*>(
          inc + (size_t)gi * NN + k0 + kseg);
      float4 a = p[0], b = p[1];
      vals[0]=a.x; vals[1]=a.y; vals[2]=a.z; vals[3]=a.w;
      vals[4]=b.x; vals[5]=b.y; vals[6]=b.z; vals[7]=b.w;
      if (k0 + 64 < NN)
        __builtin_prefetch(inc + (size_t)gi * NN + k0 + kseg + 64, 0, 1);
    }
#pragma unroll
    for (int e = 0; e < 8; ++e) {
      float v = vals[e];
      float r = v > 0.f ? v : 0.f;
      absP += fabsf(v); sqP += v * v; relP += r;
      lAr[lrow][kseg + e] = f2bf(v);
      lAl[lrow][kseg + e] = f2bf(r);
    }
    // ---- B tile: one 128B row segment per thread ----
    {
      const uint4* gp = reinterpret_cast<const uint4*>(Bt + (size_t)tid * NN + k0);
      uint4* sp = reinterpret_cast<uint4*>(&lBt[tid][0]);
#pragma unroll
      for (int c = 0; c < 8; ++c) sp[c] = gp[c];
    }
    __syncthreads();

#pragma unroll
    for (int kt = 0; kt < 2; ++kt) {
      int kb1 = kt * 32 + half * 8;
      int kb2 = kt * 32 + 16 + half * 8;
      v16bf ar = cat16(*reinterpret_cast<const v8bf*>(&lAr[mbase + mr][kb1]),
                       *reinterpret_cast<const v8bf*>(&lAr[mbase + mr][kb2]));
      v16bf al = cat16(*reinterpret_cast<const v8bf*>(&lAl[mbase + mr][kb1]),
                       *reinterpret_cast<const v8bf*>(&lAl[mbase + mr][kb2]));
      int kbB = kt * 32 + half * 16;
#pragma unroll
      for (int nt = 0; nt < 4; ++nt) {
        int n = nbase + nt * 16 + mr;
        v16bf b = cat16(*reinterpret_cast<const v8bf*>(&lBt[n][kb1]),
                        *reinterpret_cast<const v8bf*>(&lBt[n][kb2]));
        accR[nt] = wmma_bf16(ar, b, accR[nt]);
        accL[nt] = wmma_bf16(al, b, accL[nt]);
      }
    }
    __syncthreads();
  }

  // ---- epilogue: fused Frobenius diff vs master, fea(+cur) ----
  float diffP = 0.f;
#pragma unroll
  for (int nt = 0; nt < 4; ++nt) {
#pragma unroll
    for (int r = 0; r < 8; ++r) {
      int m = mbase + r + 8 * half;
      int n = nbase + nt * 16 + mr;
      size_t g = (size_t)(rowBase + m) * DD + n;
      float d = accR[nt][r] - master[g];
      diffP += d * d;
      fea[g] = accL[nt][r] + curF[g];
    }
  }

  degp[lrow][tid & 7] = relP;
  red1[tid] = absP; red2[tid] = sqP; red3[tid] = diffP;
  __syncthreads();
  if (tid < 32) {
    float s = 0.f;
#pragma unroll
    for (int q = 0; q < 8; ++q) s += degp[tid][q];
    deg[rowBase + tid] = s + 1.0f;
  }
  for (int s = 128; s > 0; s >>= 1) {
    if (tid < s) {
      red1[tid] += red1[tid + s];
      red2[tid] += red2[tid + s];
      red3[tid] += red3[tid + s];
    }
    __syncthreads();
  }
  if (tid == 0) {
    atomicAdd(&scal[SPATIAL ? 0 : 2], red1[0]);
    atomicAdd(&scal[SPATIAL ? 1 : 3], red2[0]);
    atomicAdd(&scal[SPATIAL ? 4 : 5], red3[0]);
  }
}

// ---------------------------------------------------------------------------
// Small GEMM: out[N][256] = A_bf16[N][256] @ W (opt transposed) (+bias)(+relu)
// BM=64 rows, 256 cols, 256 threads (8 waves).
// ---------------------------------------------------------------------------
__global__ __launch_bounds__(256) void gemm_nd(
    const bf16_t* __restrict__ A, const float* __restrict__ W,
    const float* __restrict__ bias, float* __restrict__ out,
    int transW, int doRelu) {
  __shared__ __align__(16) bf16_t lA[64][72];
  __shared__ __align__(16) bf16_t lBt[256][72];

  const int tid = threadIdx.x;
  const int lane = tid & 31, wave = tid >> 5;
  const int rowBase = blockIdx.x * 64;
  const int mbase = (wave & 3) * 16;
  const int nbase = (wave >> 2) * 128;
  const int mr = lane & 15, half = lane >> 4;
  const int arow = tid >> 2;
  const int akseg = (tid & 3) * 16;

  v8f zero = {0.f,0.f,0.f,0.f,0.f,0.f,0.f,0.f};
  v8f acc[8];
#pragma unroll
  for (int t = 0; t < 8; ++t) acc[t] = zero;

  for (int k0 = 0; k0 < DD; k0 += 64) {
    const uint4* ga = reinterpret_cast<const uint4*>(
        A + (size_t)(rowBase + arow) * DD + k0 + akseg);
    uint4* sa = reinterpret_cast<uint4*>(&lA[arow][akseg]);
    sa[0] = ga[0]; sa[1] = ga[1];
#pragma unroll 8
    for (int kk = 0; kk < 64; ++kk) {
      float w = transW ? W[(size_t)tid * DD + k0 + kk]
                       : W[(size_t)(k0 + kk) * DD + tid];
      lBt[tid][kk] = f2bf(w);
    }
    __syncthreads();
#pragma unroll
    for (int kt = 0; kt < 2; ++kt) {
      int kb1 = kt * 32 + half * 8, kb2 = kt * 32 + 16 + half * 8;
      v16bf a = cat16(*reinterpret_cast<const v8bf*>(&lA[mbase + mr][kb1]),
                      *reinterpret_cast<const v8bf*>(&lA[mbase + mr][kb2]));
      int kbB = kt * 32 + half * 16;
#pragma unroll
      for (int nt = 0; nt < 8; ++nt) {
        int n = nbase + nt * 16 + mr;
        v16bf b = cat16(*reinterpret_cast<const v8bf*>(&lBt[n][kb1]),
                        *reinterpret_cast<const v8bf*>(&lBt[n][kb2]));
        acc[nt] = wmma_bf16(a, b, acc[nt]);
      }
    }
    __syncthreads();
  }
#pragma unroll
  for (int nt = 0; nt < 8; ++nt) {
#pragma unroll
    for (int r = 0; r < 8; ++r) {
      int m = mbase + r + 8 * half;
      int n = nbase + nt * 16 + mr;
      float v = acc[nt][r];
      if (bias) v += bias[n];
      if (doRelu) v = v > 0.f ? v : 0.f;
      out[(size_t)(rowBase + m) * DD + n] = v;
    }
  }
}

// ---------------------------------------------------------------------------
// emb = edge_fea / deg  (written as bf16 for the next GEMM)
// ---------------------------------------------------------------------------
__global__ void emb_kernel(const float* __restrict__ spaEF, const float* __restrict__ tmpEF,
                           const float* __restrict__ sdeg, const float* __restrict__ tdeg,
                           bf16_t* __restrict__ sEmb, bf16_t* __restrict__ tEmb) {
  int idx = blockIdx.x * blockDim.x + threadIdx.x;
  if (idx >= NN * DD) return;
  int i = idx >> 8;
  sEmb[idx] = f2bf(spaEF[idx] / sdeg[i]);
  tEmb[idx] = f2bf(tmpEF[idx] / tdeg[i]);
}

// ---------------------------------------------------------------------------
// Per-row attention: val = (spa.node/16)*spa + (tmp.node/16)*tmp -> bf16
// ---------------------------------------------------------------------------
__global__ void attn_kernel(const float* __restrict__ nodeF, const float* __restrict__ spaF,
                            const float* __restrict__ tmpF, bf16_t* __restrict__ valBf) {
  __shared__ float r1[256], r2[256];
  int i = blockIdx.x, d = threadIdx.x;
  size_t g = (size_t)i * DD + d;
  float nf = nodeF[g], sf = spaF[g], tf = tmpF[g];
  r1[d] = sf * nf; r2[d] = tf * nf;
  __syncthreads();
  for (int s = 128; s > 0; s >>= 1) {
    if (d < s) { r1[d] += r1[d + s]; r2[d] += r2[d + s]; }
    __syncthreads();
  }
  float sa = r1[0] * 0.0625f, ta = r2[0] * 0.0625f;  // 1/sqrt(256)
  valBf[g] = f2bf(sa * sf + ta * tf);
}

// ---------------------------------------------------------------------------
// scalar loss finalize
// ---------------------------------------------------------------------------
__global__ void loss_kernel(const float* __restrict__ scal, float* __restrict__ out) {
  if (threadIdx.x == 0 && blockIdx.x == 0) {
    float sl = 0.2f * sqrtf(scal[4]) + scal[0] + 0.001f * sqrtf(scal[1]);
    float tl = 0.2f * sqrtf(scal[5]) + scal[2] + 0.001f * sqrtf(scal[3]);
    out[(size_t)NN * DD] = tl + sl;
  }
}

extern "C" void kernel_launch(void* const* d_in, const int* in_sizes, int n_in,
                              void* d_out, int out_size, void* d_ws, size_t ws_size,
                              hipStream_t stream) {
  (void)in_sizes; (void)n_in; (void)out_size; (void)ws_size;
  const float* cur   = (const float*)d_in[0];
  const float* pre   = (const float*)d_in[1];
  const float* rps   = (const float*)d_in[2];
  const float* incS  = (const float*)d_in[3];
  const float* rpt   = (const float*)d_in[4];
  const float* incT  = (const float*)d_in[5];
  const float* nodeP = (const float*)d_in[6];
  const float* spaP  = (const float*)d_in[7];
  const float* tmpP  = (const float*)d_in[8];
  const float* thW   = (const float*)d_in[9];
  const float* thB   = (const float*)d_in[10];
  float* out = (float*)d_out;

  char* w = (char*)d_ws;
  size_t off = 0;
  auto alloc = [&](size_t bytes) -> char* {
    char* p = w + off; off += (bytes + 255) & ~(size_t)255; return p;
  };
  const size_t ND = (size_t)NN * DD;
  float*  scal    = (float*)alloc(64 * sizeof(float));
  bf16_t* curBf   = (bf16_t*)alloc(ND * 2);
  bf16_t* curT    = (bf16_t*)alloc(ND * 2);
  bf16_t* preT    = (bf16_t*)alloc(ND * 2);
  float*  masterS = (float*)alloc(ND * 4);
  float*  masterT = (float*)alloc(ND * 4);
  float*  spaEF   = (float*)alloc(ND * 4);
  float*  tmpEF   = (float*)alloc(ND * 4);
  float*  sdeg    = (float*)alloc(NN * 4);
  float*  tdeg    = (float*)alloc(NN * 4);
  bf16_t* sEmb    = (bf16_t*)alloc(ND * 2);
  bf16_t* tEmb    = (bf16_t*)alloc(ND * 2);
  // aliases of buffers that are dead by the time these are written:
  float*  nodeFea = masterS;            // masterS consumed by big_fused<1>
  float*  spaFea  = masterT;            // masterT consumed by big_fused<0>
  float*  tmpFea  = spaEF;              // spaEF consumed by emb_kernel
  bf16_t* valBf   = (bf16_t*)tmpEF;     // tmpEF consumed by emb_kernel

  dim3 b256(256);
  prep_kernel<<<dim3(ND / 256), b256, 0, stream>>>(cur, pre, curBf, curT, preT, scal);
  gemm_nd<<<dim3(NN / 64), b256, 0, stream>>>(curBf, rps, nullptr, masterS, 0, 0);
  gemm_nd<<<dim3(NN / 64), b256, 0, stream>>>(curBf, rpt, nullptr, masterT, 0, 0);
  big_fused<1><<<dim3(NN / 32), b256, 0, stream>>>(incS, curT, cur, masterS, spaEF, sdeg, scal);
  big_fused<0><<<dim3(NN / 32), b256, 0, stream>>>(incT, preT, cur, masterT, tmpEF, tdeg, scal);
  emb_kernel<<<dim3(ND / 256), b256, 0, stream>>>(spaEF, tmpEF, sdeg, tdeg, sEmb, tEmb);
  gemm_nd<<<dim3(NN / 64), b256, 0, stream>>>(curBf, nodeP, nullptr, nodeFea, 0, 0);
  gemm_nd<<<dim3(NN / 64), b256, 0, stream>>>(sEmb, spaP, nullptr, spaFea, 0, 0);
  gemm_nd<<<dim3(NN / 64), b256, 0, stream>>>(tEmb, tmpP, nullptr, tmpFea, 0, 0);
  attn_kernel<<<dim3(NN), b256, 0, stream>>>(nodeFea, spaFea, tmpFea, valBf);
  gemm_nd<<<dim3(NN / 64), b256, 0, stream>>>(valBf, thW, thB, out, 1, 1);
  loss_kernel<<<dim3(1), dim3(1), 0, stream>>>(scal, out);
}
